// New_LinkNet_9964324126741
// MI455X (gfx1250) — compile-verified, matrix operand order checked by reading, not run on hardware
//
#include <hip/hip_runtime.h>

// ---------------------------------------------------------------------------
// CDNA5 / gfx1250 LinkNet GNN forward pass.
// All dense GEMMs run through v_wmma_f32_16x16x32_bf16 (wave32 WMMA).
// Each wave computes a 16x64 output strip (4 accumulators), loading its A
// fragment once per 32-K chunk and reusing it across 4 WMMAs.  Activations
// consumed by GEMMs are bf16 mirrors written by producer kernels; weights are
// staged transposed in LDS as bf16 (contiguous ds_read_b128 B fragments).
// ---------------------------------------------------------------------------

typedef __attribute__((ext_vector_type(16))) __bf16 v16bf;
typedef __attribute__((ext_vector_type(8)))  float  v8f;

union FragBF { v16bf v; unsigned u[8]; };

static constexpr int NNODES = 50000;
static constexpr int NEDGE  = 100000;
static constexpr int NE2    = 200000;   // bidirectional edges

__device__ __forceinline__ unsigned f2bf_bits(float x) {
  unsigned u = __float_as_uint(x);
  return (u + 0x7FFFu + ((u >> 16) & 1u)) >> 16;   // round-to-nearest-even
}
__device__ __forceinline__ unsigned short f2bf(float x) {
  return (unsigned short)f2bf_bits(x);
}

// order-preserving float<->uint encoding for atomic segment-max
__device__ __forceinline__ unsigned fenc(float f) {
  unsigned u = __float_as_uint(f);
  return (u & 0x80000000u) ? ~u : (u | 0x80000000u);
}
__device__ __forceinline__ float fdec(unsigned e) {
  unsigned u = (e & 0x80000000u) ? (e & 0x7FFFFFFFu) : ~e;
  return __uint_as_float(u);
}

__device__ __forceinline__ float apply_act(float v, int ACT) {
  if (ACT == 1) v = fmaxf(v, 0.0f);
  if (ACT == 2) v = 1.0f / (1.0f + expf(-v));
  return v;
}

// ---------------------------------------------------------------------------
// WMMA GEMM:  C[M,ND] = act(Abf[M,KD] @ W[KD,ND] + bias)
// A is bf16 row-major.  Optional row gather (rowIdx); bf16 output mirror is a
// compile-time option (MIRROR).  Block = 8 waves, each wave a 16x64 strip
// (4 accumulators, A fragment reused 4x); block covers 128 rows x 64 cols.
// ---------------------------------------------------------------------------
template <int KD, int ND, int ACT, bool MIRROR>
__global__ __launch_bounds__(256) void k_gemm(
    const unsigned short* __restrict__ A, const int* __restrict__ rowIdx,
    const float* __restrict__ W, const float* __restrict__ Bias,
    float* __restrict__ C, unsigned short* __restrict__ Cbf, int M) {
  __shared__ unsigned short sB[64 * KD];          // Bt[n][k], bf16
  const int tid = threadIdx.x;
  const int gy  = blockIdx.y;                     // which 64-column slab
  for (int i = tid; i < 64 * KD; i += 256) {
    int nloc = i / KD, kk = i % KD;
    sB[i] = f2bf(W[(size_t)kk * ND + gy * 64 + nloc]);
  }
  __syncthreads();

  const int lane = tid & 31;
  const int wid  = tid >> 5;
  const int row0 = blockIdx.x * 128 + wid * 16;
  const int l15  = lane & 15;
  const int kh   = lane >> 4;                     // K-half select (ISA layout)

  int rl = row0 + l15;
  if (rl >= M) rl = M - 1;                        // clamp tail rows (stores masked)
  if (rowIdx) rl = rowIdx[rl];
  const unsigned short* Ar = A + (size_t)rl * KD;

  v8f acc[4] = {};
#pragma unroll
  for (int k0 = 0; k0 < KD; k0 += 32) {
    FragBF fa;
    uint4 a0 = *(const uint4*)(Ar + k0 + kh * 8);        // 8 bf16
    uint4 a1 = *(const uint4*)(Ar + k0 + 16 + kh * 8);   // 8 bf16
    fa.u[0] = a0.x; fa.u[1] = a0.y; fa.u[2] = a0.z; fa.u[3] = a0.w;
    fa.u[4] = a1.x; fa.u[5] = a1.y; fa.u[6] = a1.z; fa.u[7] = a1.w;
#pragma unroll
    for (int nt = 0; nt < 4; ++nt) {
      FragBF fb;
      uint4 b0 = *(const uint4*)&sB[(nt * 16 + l15) * KD + k0 + kh * 8];
      uint4 b1 = *(const uint4*)&sB[(nt * 16 + l15) * KD + k0 + 16 + kh * 8];
      fb.u[0] = b0.x; fb.u[1] = b0.y; fb.u[2] = b0.z; fb.u[3] = b0.w;
      fb.u[4] = b1.x; fb.u[5] = b1.y; fb.u[6] = b1.z; fb.u[7] = b1.w;
      acc[nt] = __builtin_amdgcn_wmma_f32_16x16x32_bf16(
          false, fa.v, false, fb.v, (short)0, acc[nt], false, false);
    }
  }

  if (row0 + 16 <= M) {                           // full tile: branch-free stores
#pragma unroll
    for (int nt = 0; nt < 4; ++nt) {
      const int colg = gy * 64 + nt * 16 + l15;
      const float bb = Bias[colg];
#pragma unroll
      for (int r = 0; r < 8; ++r) {
        float v = apply_act(acc[nt][r] + bb, ACT);
        size_t o = (size_t)(row0 + kh * 8 + r) * ND + colg;
        C[o] = v;
        if (MIRROR) Cbf[o] = f2bf(v);
      }
    }
  } else {
#pragma unroll
    for (int nt = 0; nt < 4; ++nt) {
      const int colg = gy * 64 + nt * 16 + l15;
      const float bb = Bias[colg];
#pragma unroll
      for (int r = 0; r < 8; ++r) {
        int orow = row0 + kh * 8 + r;
        if (orow < M) {
          float v = apply_act(acc[nt][r] + bb, ACT);
          size_t o = (size_t)orow * ND + colg;
          C[o] = v;
          if (MIRROR) Cbf[o] = f2bf(v);
        }
      }
    }
  }
}

// ---------------------------------------------------------------------------
// WMMA GEMM on concatenated A = [gather(A0,idx0) | gather(A1,idx1)], K=128
// (two 64-wide bf16 halves), N=64.  Gating MLPs and the edge classifier.
// Same 16x64-strip wave mapping as k_gemm.
// ---------------------------------------------------------------------------
template <int ACT>
__global__ __launch_bounds__(256) void k_gemm_cat(
    const unsigned short* __restrict__ A0, const int* __restrict__ idx0,
    const unsigned short* __restrict__ A1, const int* __restrict__ idx1,
    const float* __restrict__ W, const float* __restrict__ Bias,
    float* __restrict__ C, int M) {
  __shared__ unsigned short sB[64 * 128];
  const int tid = threadIdx.x;
  for (int i = tid; i < 64 * 128; i += 256) {
    int nloc = i / 128, kk = i % 128;
    sB[i] = f2bf(W[(size_t)kk * 64 + nloc]);
  }
  __syncthreads();

  const int lane = tid & 31;
  const int wid  = tid >> 5;
  const int row0 = blockIdx.x * 128 + wid * 16;
  const int l15  = lane & 15;
  const int kh   = lane >> 4;

  int rl = row0 + l15;
  if (rl >= M) rl = M - 1;
  const int r0 = idx0 ? idx0[rl] : rl;
  const int r1 = idx1 ? idx1[rl] : rl;
  const unsigned short* A0r = A0 + (size_t)r0 * 64;
  const unsigned short* A1r = A1 + (size_t)r1 * 64;

  v8f acc[4] = {};
#pragma unroll
  for (int k0 = 0; k0 < 128; k0 += 32) {
    const unsigned short* Ar = (k0 < 64) ? A0r : A1r;
    const int koff = k0 & 63;
    FragBF fa;
    uint4 a0 = *(const uint4*)(Ar + koff + kh * 8);
    uint4 a1 = *(const uint4*)(Ar + koff + 16 + kh * 8);
    fa.u[0] = a0.x; fa.u[1] = a0.y; fa.u[2] = a0.z; fa.u[3] = a0.w;
    fa.u[4] = a1.x; fa.u[5] = a1.y; fa.u[6] = a1.z; fa.u[7] = a1.w;
#pragma unroll
    for (int nt = 0; nt < 4; ++nt) {
      FragBF fb;
      uint4 b0 = *(const uint4*)&sB[(nt * 16 + l15) * 128 + k0 + kh * 8];
      uint4 b1 = *(const uint4*)&sB[(nt * 16 + l15) * 128 + k0 + 16 + kh * 8];
      fb.u[0] = b0.x; fb.u[1] = b0.y; fb.u[2] = b0.z; fb.u[3] = b0.w;
      fb.u[4] = b1.x; fb.u[5] = b1.y; fb.u[6] = b1.z; fb.u[7] = b1.w;
      acc[nt] = __builtin_amdgcn_wmma_f32_16x16x32_bf16(
          false, fa.v, false, fb.v, (short)0, acc[nt], false, false);
    }
  }

  if (row0 + 16 <= M) {
#pragma unroll
    for (int nt = 0; nt < 4; ++nt) {
      const int colg = nt * 16 + l15;
      const float bb = Bias[colg];
#pragma unroll
      for (int r = 0; r < 8; ++r) {
        float v = apply_act(acc[nt][r] + bb, ACT);
        C[(size_t)(row0 + kh * 8 + r) * 64 + colg] = v;
      }
    }
  } else {
#pragma unroll
    for (int nt = 0; nt < 4; ++nt) {
      const int colg = nt * 16 + l15;
      const float bb = Bias[colg];
#pragma unroll
      for (int r = 0; r < 8; ++r) {
        int orow = row0 + kh * 8 + r;
        if (orow < M)
          C[(size_t)orow * 64 + colg] = apply_act(acc[nt][r] + bb, ACT);
      }
    }
  }
}

// ------------------------- elementwise / graph kernels ---------------------

__global__ void k_fill(float* __restrict__ p, float v, int n) {
  int i = blockIdx.x * 256 + threadIdx.x;
  if (i < n) p[i] = v;
}

__global__ void k_build_edges(const int* __restrict__ ei, int* __restrict__ s,
                              int* __restrict__ d) {
  int i = blockIdx.x * 256 + threadIdx.x;
  if (i >= NE2) return;
  if (i < NEDGE) { s[i] = ei[i];  d[i] = ei[NEDGE + i]; }
  else           { s[i] = ei[i];  d[i] = ei[i - NEDGE]; }
}

__global__ void k_embed_node(const float* __restrict__ x, const float* __restrict__ Wn,
                             const float* __restrict__ bn, const float* __restrict__ zt,
                             const int* __restrict__ zk, float* __restrict__ nf) {
  int i = blockIdx.x * 256 + threadIdx.x;
  if (i >= NNODES * 64) return;
  int n = i >> 6, c = i & 63;
  float a = bn[c];
#pragma unroll
  for (int j = 0; j < 6; ++j) a += x[n * 6 + j] * Wn[j * 64 + c];
  a = fmaxf(a, 0.0f);
  int zi = (int)rintf(x[n * 6 + 2]);
  int idx = 0;
#pragma unroll
  for (int j = 5; j >= 0; --j) if (zi == zk[j]) idx = j;  // first match wins
  nf[i] = a + zt[idx * 64 + c];
}

__global__ void k_embed_edge(const float* __restrict__ ea, const float* __restrict__ x,
                             const int* __restrict__ sA, const int* __restrict__ dA,
                             const float* __restrict__ We, const float* __restrict__ be,
                             float* __restrict__ ef) {
  int i = blockIdx.x * 256 + threadIdx.x;
  if (i >= NE2 * 64) return;
  int e = i >> 6, c = i & 63;
  int e0 = (e < NEDGE) ? e : e - NEDGE;
  int s = sA[e], d = dA[e];
  float a = be[c];
#pragma unroll
  for (int j = 0; j < 4; ++j) a += ea[e0 * 4 + j] * We[j * 64 + c];
#pragma unroll
  for (int j = 0; j < 3; ++j)
    a += (x[s * 6 + j] - x[d * 6 + j]) * We[(4 + j) * 64 + c];
  ef[i] = fmaxf(a, 0.0f);
}

// ---- CBAM ----
__global__ void k_colreduce(const float* __restrict__ X, int R, float* __restrict__ stats) {
  __shared__ float smax[256], ssum[256];
  int c = blockIdx.x, t = threadIdx.x;
  float mx = -3.4e38f, sm = 0.0f;
  for (int r = t; r < R; r += 256) {
    float v = X[(size_t)r * 64 + c];
    mx = fmaxf(mx, v); sm += v;
  }
  smax[t] = mx; ssum[t] = sm; __syncthreads();
  for (int o = 128; o; o >>= 1) {
    if (t < o) { smax[t] = fmaxf(smax[t], smax[t + o]); ssum[t] += ssum[t + o]; }
    __syncthreads();
  }
  if (t == 0) { stats[c] = smax[0]; stats[64 + c] = ssum[0] / (float)R; }
}

__global__ void k_chmlp(const float* __restrict__ stats, const float* __restrict__ w1,
                        const float* __restrict__ b1, const float* __restrict__ w2,
                        const float* __restrict__ b2, float* __restrict__ cw) {
  __shared__ float hid[4];
  int t = threadIdx.x;   // 64 threads
  if (t < 4) {
    float a = b1[t];
    for (int i = 0; i < 128; ++i) a += stats[i] * w1[i * 4 + t];
    hid[t] = fmaxf(a, 0.0f);
  }
  __syncthreads();
  float a = b2[t];
#pragma unroll
  for (int r = 0; r < 4; ++r) a += hid[r] * w2[r * 64 + t];
  cw[t] = 1.0f / (1.0f + expf(-a));
}

__global__ void k_rowscale_pool(float* __restrict__ X, const float* __restrict__ cw,
                                int R, float* __restrict__ sp0, float* __restrict__ sp1) {
  int r = blockIdx.x * 256 + threadIdx.x;
  if (r >= R) return;
  float mx = -3.4e38f, sm = 0.0f;
#pragma unroll 4
  for (int c = 0; c < 64; ++c) {
    float v = X[(size_t)r * 64 + c] * cw[c];
    X[(size_t)r * 64 + c] = v;
    mx = fmaxf(mx, v); sm += v;
  }
  sp0[r] = mx; sp1[r] = sm * (1.0f / 64.0f);
}

__global__ void k_conv1(const float* __restrict__ sp0, const float* __restrict__ sp1,
                        const float* __restrict__ w, const float* __restrict__ b,
                        float* __restrict__ s1, int R) {
  int r = blockIdx.x * 256 + threadIdx.x;
  if (r >= R) return;
  float a = b[0];
#pragma unroll
  for (int j = 0; j < 5; ++j) {
    int rr = r + j - 2;
    if (rr >= 0 && rr < R) a += w[j] * sp0[rr] + w[5 + j] * sp1[rr];
  }
  s1[r] = fmaxf(a, 0.0f);
}

// final CBAM scale; also writes the bf16 mirror used by the WMMA GEMMs
__global__ void k_conv2_scale(float* __restrict__ X, unsigned short* __restrict__ Xbf,
                              const float* __restrict__ s1, const float* __restrict__ w,
                              const float* __restrict__ b, int R) {
  int r = blockIdx.x * 256 + threadIdx.x;
  if (r >= R) return;
  float a = b[0];
#pragma unroll
  for (int j = 0; j < 3; ++j) {
    int rr = r + j - 1;
    if (rr >= 0 && rr < R) a += w[j] * s1[rr];
  }
  float s = 1.0f / (1.0f + expf(-a));
#pragma unroll 4
  for (int c = 0; c < 64; ++c) {
    float v = X[(size_t)r * 64 + c] * s;
    X[(size_t)r * 64 + c] = v;
    Xbf[(size_t)r * 64 + c] = f2bf(v);
  }
}

// ---- message passing ----
__global__ void k_segsum(const float* __restrict__ ef, const int* __restrict__ dA,
                         float* __restrict__ agg) {
  int i = blockIdx.x * 256 + threadIdx.x;
  if (i >= NE2 * 64) return;
  int e = i >> 6, c = i & 63;
  atomicAdd(&agg[(size_t)dA[e] * 64 + c], ef[i]);
}

// LayerNorm over concat([nf, agg-nf]) (D=128) -> bf16, one wave32 per row
__global__ __launch_bounds__(256) void k_ln_comb(
    const float* __restrict__ nf, const float* __restrict__ agg,
    const float* __restrict__ G, const float* __restrict__ Bb,
    unsigned short* __restrict__ Y, int Rows) {
  int wid = threadIdx.x >> 5, lane = threadIdx.x & 31;
  int row = blockIdx.x * 8 + wid;
  if (row >= Rows) return;
  float v[4], s = 0.0f;
#pragma unroll
  for (int j = 0; j < 4; ++j) {
    int c = lane + 32 * j;
    float a;
    if (c < 64) a = nf[(size_t)row * 64 + c];
    else        a = agg[(size_t)row * 64 + (c - 64)] - nf[(size_t)row * 64 + (c - 64)];
    v[j] = a; s += a;
  }
  for (int o = 16; o; o >>= 1) s += __shfl_xor(s, o, 32);
  float mu = s * (1.0f / 128.0f), vs = 0.0f;
#pragma unroll
  for (int j = 0; j < 4; ++j) { float d = v[j] - mu; vs += d * d; }
  for (int o = 16; o; o >>= 1) vs += __shfl_xor(vs, o, 32);
  float inv = rsqrtf(vs * (1.0f / 128.0f) + 1e-5f);
#pragma unroll
  for (int j = 0; j < 4; ++j) {
    int c = lane + 32 * j;
    Y[(size_t)row * 128 + c] = f2bf((v[j] - mu) * inv * G[c] + Bb[c]);
  }
}

// LayerNorm D=256 -> bf16 (GEMM-only consumer)
__global__ __launch_bounds__(256) void k_ln256(
    const float* __restrict__ X, const float* __restrict__ G,
    const float* __restrict__ Bb, unsigned short* __restrict__ Y, int Rows) {
  int wid = threadIdx.x >> 5, lane = threadIdx.x & 31;
  int row = blockIdx.x * 8 + wid;
  if (row >= Rows) return;
  float v[8], s = 0.0f;
#pragma unroll
  for (int j = 0; j < 8; ++j) { v[j] = X[(size_t)row * 256 + lane + 32 * j]; s += v[j]; }
  for (int o = 16; o; o >>= 1) s += __shfl_xor(s, o, 32);
  float mu = s * (1.0f / 256.0f), vs = 0.0f;
#pragma unroll
  for (int j = 0; j < 8; ++j) { float d = v[j] - mu; vs += d * d; }
  for (int o = 16; o; o >>= 1) vs += __shfl_xor(vs, o, 32);
  float inv = rsqrtf(vs * (1.0f / 256.0f) + 1e-5f);
#pragma unroll
  for (int j = 0; j < 8; ++j) {
    int c = lane + 32 * j;
    Y[(size_t)row * 256 + c] = f2bf((v[j] - mu) * inv * G[c] + Bb[c]);
  }
}

__global__ void k_logits(const float* __restrict__ qb, const float* __restrict__ kb,
                         const int* __restrict__ sA, const int* __restrict__ dA,
                         float* __restrict__ lg, unsigned* __restrict__ menc) {
  int i = blockIdx.x * 256 + threadIdx.x;
  if (i >= NE2 * 4) return;
  int e = i >> 2, h = i & 3;
  const float* qr = qb + (size_t)dA[e] * 256 + h * 64;
  const float* kr = kb + (size_t)sA[e] * 256 + h * 64;
  float a = 0.0f;
#pragma unroll 8
  for (int c = 0; c < 64; ++c) a += qr[c] * kr[c];
  a *= 0.125f;                                   // 1/sqrt(64)
  lg[i] = a;
  atomicMax(&menc[(size_t)dA[e] * 4 + h], fenc(a));
}

__global__ void k_pden(float* __restrict__ lg, const unsigned* __restrict__ menc,
                       const int* __restrict__ dA, float* __restrict__ denom) {
  int i = blockIdx.x * 256 + threadIdx.x;
  if (i >= NE2 * 4) return;
  int e = i >> 2, h = i & 3;
  float m = fdec(menc[(size_t)dA[e] * 4 + h]);
  float p = expf(lg[i] - m);
  lg[i] = p;
  atomicAdd(&denom[(size_t)dA[e] * 4 + h], p);
}

__global__ void k_attacc(const float* __restrict__ p4, const float* __restrict__ vb,
                         const int* __restrict__ sA, const int* __restrict__ dA,
                         float* __restrict__ att) {
  int i = blockIdx.x * 256 + threadIdx.x;
  if (i >= NE2 * 256) return;
  int e = i >> 8, hc = i & 255, h = hc >> 6;
  float p = p4[(size_t)e * 4 + h];
  atomicAdd(&att[(size_t)dA[e] * 256 + hc], p * vb[(size_t)sA[e] * 256 + hc]);
}

__global__ void k_attfin(float* __restrict__ att, const float* __restrict__ denom,
                         const float* __restrict__ skp) {
  int i = blockIdx.x * 256 + threadIdx.x;
  if (i >= NNODES * 256) return;
  int n = i >> 8, h = (i >> 6) & 3;
  float dn = denom[(size_t)n * 4 + h];
  float o = (dn > 0.0f) ? att[i] / dn : 0.0f;
  att[i] = o + skp[i];
}

// gated mix; writes f32 (elementwise consumers) + bf16 mirror (GEMM consumers)
__global__ void k_mix(const float* __restrict__ x0, const float* __restrict__ pj,
                      const float* __restrict__ g, float* __restrict__ out,
                      unsigned short* __restrict__ outbf, int n) {
  int i = blockIdx.x * 256 + threadIdx.x;
  if (i >= n) return;
  float gg = g[i];
  float v = x0[i] * gg + pj[i] * (1.0f - gg);
  out[i] = v;
  outbf[i] = f2bf(v);
}

__global__ void k_final(const float* __restrict__ h, const float* __restrict__ w2,
                        const float* __restrict__ b2, float* __restrict__ out) {
  int e = blockIdx.x * 256 + threadIdx.x;
  if (e >= NEDGE) return;
  float a = b2[0];
#pragma unroll 8
  for (int c = 0; c < 64; ++c) a += h[(size_t)e * 64 + c] * w2[c];
  out[e] = a;
}

// ---------------------------------------------------------------------------
static inline int cdiv(int a, int b) { return (a + b - 1) / b; }

static void run_cbam(float* feat, unsigned short* featbf, int R,
                     const float* w1, const float* b1, const float* w2, const float* b2,
                     const float* cw1, const float* cb1, const float* cw2, const float* cb2,
                     float* stats, float* cw, float* sp0, float* sp1, float* s1,
                     hipStream_t stream) {
  k_colreduce<<<64, 256, 0, stream>>>(feat, R, stats);
  k_chmlp<<<1, 64, 0, stream>>>(stats, w1, b1, w2, b2, cw);
  k_rowscale_pool<<<cdiv(R, 256), 256, 0, stream>>>(feat, cw, R, sp0, sp1);
  k_conv1<<<cdiv(R, 256), 256, 0, stream>>>(sp0, sp1, cw1, cb1, s1, R);
  k_conv2_scale<<<cdiv(R, 256), 256, 0, stream>>>(feat, featbf, s1, cw2, cb2, R);
}

extern "C" void kernel_launch(void* const* d_in, const int* in_sizes, int n_in,
                              void* d_out, int out_size, void* d_ws, size_t ws_size,
                              hipStream_t stream) {
  (void)in_sizes; (void)n_in; (void)out_size; (void)ws_size;
  const float* x         = (const float*)d_in[0];
  const float* edge_attr = (const float*)d_in[1];
  const int*   edge_idx  = (const int*)d_in[2];
  const int*   z_keys    = (const int*)d_in[3];
  const float* z_table   = (const float*)d_in[4];
  const float* W_node    = (const float*)d_in[5];
  const float* b_node    = (const float*)d_in[6];
  const float* W_edge    = (const float*)d_in[7];
  const float* b_edge    = (const float*)d_in[8];
  const float* cn_w1 = (const float*)d_in[9],  *cn_b1 = (const float*)d_in[10];
  const float* cn_w2 = (const float*)d_in[11], *cn_b2 = (const float*)d_in[12];
  const float* cn_cw1 = (const float*)d_in[13], *cn_cb1 = (const float*)d_in[14];
  const float* cn_cw2 = (const float*)d_in[15], *cn_cb2 = (const float*)d_in[16];
  const float* ce_w1 = (const float*)d_in[17], *ce_b1 = (const float*)d_in[18];
  const float* ce_w2 = (const float*)d_in[19], *ce_b2 = (const float*)d_in[20];
  const float* ce_cw1 = (const float*)d_in[21], *ce_cb1 = (const float*)d_in[22];
  const float* ce_cw2 = (const float*)d_in[23], *ce_cb2 = (const float*)d_in[24];
  const float* ln_comb_g = (const float*)d_in[25], *ln_comb_b = (const float*)d_in[26];
  const float* Wq = (const float*)d_in[27], *bq = (const float*)d_in[28];
  const float* Wk = (const float*)d_in[29], *bk = (const float*)d_in[30];
  const float* Wv = (const float*)d_in[31], *bv = (const float*)d_in[32];
  const float* Ws = (const float*)d_in[33], *bs = (const float*)d_in[34];
  const float* ln_tc_g = (const float*)d_in[35], *ln_tc_b = (const float*)d_in[36];
  const float* Wpn = (const float*)d_in[37], *bpn = (const float*)d_in[38];
  const float* Wpe = (const float*)d_in[39], *bpe = (const float*)d_in[40];
  const float* Wg = (const float*)d_in[41], *bg = (const float*)d_in[42];
  const float* Wc1 = (const float*)d_in[43], *bc1 = (const float*)d_in[44];
  const float* Wc2 = (const float*)d_in[45], *bc2 = (const float*)d_in[46];

  // ---- bump allocator over d_ws ----
  char* wp = (char*)d_ws;
  size_t off = 0;
  auto alloc = [&](size_t bytes) -> void* {
    void* p = wp + off;
    off += (bytes + 255) & ~(size_t)255;
    return p;
  };
  int*   srcA = (int*)alloc((size_t)NE2 * 4);
  int*   dstA = (int*)alloc((size_t)NE2 * 4);
  float* nfA  = (float*)alloc((size_t)NNODES * 64 * 4);
  float* nfB  = (float*)alloc((size_t)NNODES * 64 * 4);
  float* efA  = (float*)alloc((size_t)NE2 * 64 * 4);
  float* efB  = (float*)alloc((size_t)NE2 * 64 * 4);
  unsigned short* nfAbf = (unsigned short*)alloc((size_t)NNODES * 64 * 2);
  unsigned short* nfBbf = (unsigned short*)alloc((size_t)NNODES * 64 * 2);
  unsigned short* efAbf = (unsigned short*)alloc((size_t)NE2 * 64 * 2);
  unsigned short* efBbf = (unsigned short*)alloc((size_t)NE2 * 64 * 2);
  float* agg  = (float*)alloc((size_t)NNODES * 64 * 4);
  unsigned short* combbf = (unsigned short*)alloc((size_t)NNODES * 128 * 2);
  float* qb   = (float*)alloc((size_t)NNODES * 256 * 4);
  float* kb   = (float*)alloc((size_t)NNODES * 256 * 4);
  float* vb   = (float*)alloc((size_t)NNODES * 256 * 4);
  float* skp  = (float*)alloc((size_t)NNODES * 256 * 4);
  float* lg   = (float*)alloc((size_t)NE2 * 4 * 4);
  unsigned* menc = (unsigned*)alloc((size_t)NNODES * 4 * 4);
  float* denom = (float*)alloc((size_t)NNODES * 4 * 4);
  float* att  = (float*)alloc((size_t)NNODES * 256 * 4);
  unsigned short* toutbf = (unsigned short*)alloc((size_t)NNODES * 256 * 2);
  float* npj  = (float*)alloc((size_t)NNODES * 64 * 4);
  float* epj  = (float*)alloc((size_t)NE2 * 64 * 4);
  unsigned short* npjbf = (unsigned short*)alloc((size_t)NNODES * 64 * 2);
  unsigned short* epjbf = (unsigned short*)alloc((size_t)NE2 * 64 * 2);
  float* gn   = (float*)alloc((size_t)NNODES * 64 * 4);
  float* ge   = (float*)alloc((size_t)NE2 * 64 * 4);
  float* clfh = (float*)alloc((size_t)NEDGE * 64 * 4);
  float* sp0  = (float*)alloc((size_t)NE2 * 4);
  float* sp1  = (float*)alloc((size_t)NE2 * 4);
  float* s1b  = (float*)alloc((size_t)NE2 * 4);
  float* stats = (float*)alloc(128 * 4);
  float* cwb   = (float*)alloc(64 * 4);

  // ---- embeddings + CBAM ----
  k_build_edges<<<cdiv(NE2, 256), 256, 0, stream>>>(edge_idx, srcA, dstA);
  k_embed_node<<<cdiv(NNODES * 64, 256), 256, 0, stream>>>(x, W_node, b_node,
                                                           z_table, z_keys, nfA);
  run_cbam(nfA, nfAbf, NNODES, cn_w1, cn_b1, cn_w2, cn_b2,
           cn_cw1, cn_cb1, cn_cw2, cn_cb2, stats, cwb, sp0, sp1, s1b, stream);
  k_embed_edge<<<cdiv(NE2 * 64, 256), 256, 0, stream>>>(edge_attr, x, srcA, dstA,
                                                        W_edge, b_edge, efA);
  run_cbam(efA, efAbf, NE2, ce_w1, ce_b1, ce_w2, ce_b2,
           ce_cw1, ce_cb1, ce_cw2, ce_cb2, stats, cwb, sp0, sp1, s1b, stream);

  // ---- message passing iterations ----
  float* nf = nfA; float* nfN = nfB;
  float* ef = efA; float* efN = efB;
  unsigned short* nfbf = nfAbf; unsigned short* nfNbf = nfBbf;
  unsigned short* efbf = efAbf; unsigned short* efNbf = efBbf;
  for (int t = 0; t < 3; ++t) {
    k_fill<<<cdiv(NNODES * 64, 256), 256, 0, stream>>>(agg, 0.0f, NNODES * 64);
    k_segsum<<<cdiv(NE2 * 64, 256), 256, 0, stream>>>(ef, dstA, agg);
    k_ln_comb<<<cdiv(NNODES, 8), 256, 0, stream>>>(nf, agg, ln_comb_g + t * 128,
                                                   ln_comb_b + t * 128, combbf, NNODES);
    dim3 g256(cdiv(NNODES, 128), 4);
    k_gemm<128, 256, 0, false><<<g256, 256, 0, stream>>>(
        combbf, nullptr, Wq + t * 128 * 256, bq + t * 256, qb, nullptr, NNODES);
    k_gemm<128, 256, 0, false><<<g256, 256, 0, stream>>>(
        combbf, nullptr, Wk + t * 128 * 256, bk + t * 256, kb, nullptr, NNODES);
    k_gemm<128, 256, 0, false><<<g256, 256, 0, stream>>>(
        combbf, nullptr, Wv + t * 128 * 256, bv + t * 256, vb, nullptr, NNODES);
    k_gemm<128, 256, 0, false><<<g256, 256, 0, stream>>>(
        combbf, nullptr, Ws + t * 128 * 256, bs + t * 256, skp, nullptr, NNODES);
    k_fill<<<cdiv(NNODES * 4, 256), 256, 0, stream>>>((float*)menc, 0.0f, NNODES * 4);
    k_fill<<<cdiv(NNODES * 4, 256), 256, 0, stream>>>(denom, 0.0f, NNODES * 4);
    k_logits<<<cdiv(NE2 * 4, 256), 256, 0, stream>>>(qb, kb, srcA, dstA, lg, menc);
    k_pden<<<cdiv(NE2 * 4, 256), 256, 0, stream>>>(lg, menc, dstA, denom);
    k_fill<<<cdiv(NNODES * 256, 256), 256, 0, stream>>>(att, 0.0f, NNODES * 256);
    k_attacc<<<cdiv(NE2 * 256, 256), 256, 0, stream>>>(lg, vb, srcA, dstA, att);
    k_attfin<<<cdiv(NNODES * 256, 256), 256, 0, stream>>>(att, denom, skp);
    k_ln256<<<cdiv(NNODES, 8), 256, 0, stream>>>(att, ln_tc_g + t * 256,
                                                 ln_tc_b + t * 256, toutbf, NNODES);
    k_gemm<256, 64, 0, true><<<dim3(cdiv(NNODES, 128), 1), 256, 0, stream>>>(
        toutbf, nullptr, Wpn + t * 256 * 64, bpn + t * 64, npj, npjbf, NNODES);
    k_gemm<256, 64, 0, true><<<dim3(cdiv(NE2, 128), 1), 256, 0, stream>>>(
        toutbf, srcA, Wpe + t * 256 * 64, bpe + t * 64, epj, epjbf, NE2);
    k_gemm_cat<2><<<dim3(cdiv(NNODES, 128), 1), 256, 0, stream>>>(
        nfbf, nullptr, npjbf, nullptr, Wg + t * 128 * 64, bg + t * 64, gn, NNODES);
    k_gemm_cat<2><<<dim3(cdiv(NE2, 128), 1), 256, 0, stream>>>(
        efbf, nullptr, epjbf, nullptr, Wg + t * 128 * 64, bg + t * 64, ge, NE2);
    k_mix<<<cdiv(NNODES * 64, 256), 256, 0, stream>>>(nf, npj, gn, nfN, nfNbf,
                                                      NNODES * 64);
    k_mix<<<cdiv(NE2 * 64, 256), 256, 0, stream>>>(ef, epj, ge, efN, efNbf,
                                                   NE2 * 64);
    float* tf;
    unsigned short* tb;
    tf = nf; nf = nfN; nfN = tf;   tb = nfbf; nfbf = nfNbf; nfNbf = tb;
    tf = ef; ef = efN; efN = tf;   tb = efbf; efbf = efNbf; efNbf = tb;
  }

  // ---- edge classifier ----
  k_gemm_cat<1><<<dim3(cdiv(NEDGE, 128), 1), 256, 0, stream>>>(
      nfbf, edge_idx, nfbf, edge_idx + NEDGE, Wc1, bc1, clfh, NEDGE);
  k_final<<<cdiv(NEDGE, 256), 256, 0, stream>>>(clfh, Wc2, bc2, (float*)d_out);
}
